// InstantNGP_50440095924764
// MI455X (gfx1250) — matrix operand order repaired and checked by reading, not compiled, not run on hardware
//
#include <hip/hip_runtime.h>
#include <hip/hip_bf16.h>

typedef __attribute__((ext_vector_type(16))) _Float16 v16h;
typedef __attribute__((ext_vector_type(8)))  float    v8f;

#define NGP_T        (1u << 19)       // hash table entries per level
#define ROW_H        72               // halves per activation row (144 B, 16B aligned, conflict-free)
#define WAVE_ROWS    16
#define NFRAGS       36               // total WMMA B fragments across 7 layers
#define NFRAG_HALVES (NFRAGS * 512)   // 18432 halves = 36864 bytes
#define NBIAS        352              // padded bias floats (64,64,16,64,64,64,16)

struct WPtrs {
  const float* W[7];
  const float* b[7];
};

// ---------------------------------------------------------------------------
// Pre-pack: f32 row-major weights -> f16 WMMA B-fragments (contiguous-k B 32x16
// layout: lanes 0-15 hold K=0..15 (k=e), lanes 16-31 hold K=16..31, col=lane&15)
// plus zero-padded biases. Output into d_ws.
// ---------------------------------------------------------------------------
__global__ __launch_bounds__(256) void ngp_pack(WPtrs P, _Float16* wsW, float* wsB) {
  int idx = blockIdx.x * 256 + threadIdx.x;
  const int fragStart[8] = {0, 4, 12, 14, 18, 26, 34, 36};
  const int ntiles[7]    = {4, 4, 1, 4, 4, 4, 1};
  const int kreal[7]     = {32, 64, 64, 31, 64, 64, 64};
  const int nreal[7]     = {64, 64, 16, 64, 64, 64, 3};
  if (idx < NFRAG_HALVES) {
    int frag = idx >> 9;          // fragment index (512 halves each)
    int r    = idx & 511;
    int lane = r >> 4;
    int e    = r & 15;
    int li = 6;
    for (int i = 0; i < 7; ++i) { if (frag < fragStart[i + 1]) { li = i; break; } }
    int local = frag - fragStart[li];
    int nt = ntiles[li];
    int kk = local / nt;
    int n  = local - kk * nt;
    int ncol = n * 16 + (lane & 15);
    int k = kk * 32 + ((lane >> 4) << 4) + e;
    float v = 0.0f;
    if (k < kreal[li] && ncol < nreal[li]) v = P.W[li][k * nreal[li] + ncol];
    wsW[idx] = (_Float16)v;
  } else if (idx < NFRAG_HALVES + NBIAS) {
    int j = idx - NFRAG_HALVES;
    const int boff[7] = {0, 64, 128, 144, 208, 272, 336};
    const int bpad[7] = {64, 64, 16, 64, 64, 64, 16};
    const int brl[7]  = {64, 64, 16, 64, 64, 64, 3};
    int li = 6;
    for (int i = 0; i < 7; ++i) { if (j < boff[i] + bpad[i]) { li = i; break; } }
    int o = j - boff[li];
    wsB[j] = (o < brl[li]) ? P.b[li][o] : 0.0f;
  }
}

// ---------------------------------------------------------------------------
// WMMA fragment loads from LDS
// ---------------------------------------------------------------------------
// A fragment (A 16x32, contiguous k, same as B): lanes 0-15: M=lane, halves cover K
// {32kk..32kk+15}; lanes 16-31: M=lane-16, K offset +16.
__device__ __forceinline__ v16h ldsA(const _Float16* act, int lane, int kk) {
  int m  = lane & 15;
  int ko = kk * 32 + ((lane >> 4) << 4);
  const _Float16* p = act + m * ROW_H + ko;
  union { uint4 u[2]; v16h v; } un;
  un.u[0] = *reinterpret_cast<const uint4*>(p);        // 8 halves
  un.u[1] = *reinterpret_cast<const uint4*>(p + 8);    // 8 halves, +16B
  return un.v;
}

// B fragment: pre-packed contiguous 32 B per lane.
__device__ __forceinline__ v16h ldsB(const _Float16* sW, int frag, int lane) {
  const _Float16* p = sW + frag * 512 + lane * 16;
  union { uint4 u[2]; v16h v; } un;
  un.u[0] = *reinterpret_cast<const uint4*>(p);
  un.u[1] = *reinterpret_cast<const uint4*>(p + 8);
  return un.v;
}

// One dense layer for this wave's 16-row tile: K = 32*KS, Nout = 16*NT.
template <int KS, int NT>
__device__ __forceinline__ void mlp_layer(const _Float16* act, const _Float16* sW,
                                          const float* bias, int fragBase, int lane,
                                          v8f out[NT]) {
  v16h a[KS];
  #pragma unroll
  for (int kk = 0; kk < KS; ++kk) a[kk] = ldsA(act, lane, kk);
  #pragma unroll
  for (int n = 0; n < NT; ++n) {
    v8f c = {};
    #pragma unroll
    for (int kk = 0; kk < KS; ++kk) {
      v16h b = ldsB(sW, fragBase + kk * NT + n, lane);
      c = __builtin_amdgcn_wmma_f32_16x16x32_f16(false, a[kk], false, b,
                                                 (short)0, c, false, false);
    }
    float bb = bias[n * 16 + (lane & 15)];
    #pragma unroll
    for (int r = 0; r < 8; ++r) c[r] += bb;
    out[n] = c;
  }
}

// ReLU + f16 store of D fragments back to the activation tile.
// D layout: VGPR r -> row = 8*(lane>=16)+r, col = ntile*16 + (lane&15).
template <int NT>
__device__ __forceinline__ void store_relu(_Float16* act, int lane, const v8f* c) {
  int col0 = lane & 15;
  int rb   = (lane >> 4) << 3;
  #pragma unroll
  for (int n = 0; n < NT; ++n)
    #pragma unroll
    for (int r = 0; r < 8; ++r)
      act[(rb + r) * ROW_H + n * 16 + col0] = (_Float16)fmaxf(c[n][r], 0.0f);
}

// ---------------------------------------------------------------------------
// Main fused kernel: 128 points / block (8 waves x 16-row WMMA tiles)
// ---------------------------------------------------------------------------
__global__ __launch_bounds__(256) void ngp_fwd(const float* __restrict__ points,
                                               const float* __restrict__ views,
                                               const float* __restrict__ tables,
                                               const _Float16* __restrict__ wsPack,
                                               float* __restrict__ out, int N) {
  __shared__ _Float16 sW[NFRAG_HALVES];                 // 36864 B packed weights
  __shared__ float    sB[NBIAS];                        // 1408 B biases
  __shared__ _Float16 sAct[8 * WAVE_ROWS * ROW_H];      // 18432 B activations

  const int tid = threadIdx.x;

  // ---- Stage packed weights + biases into LDS (b128 copies) ----
  {
    const uint4* src  = reinterpret_cast<const uint4*>(wsPack);
    uint4*       dstW = reinterpret_cast<uint4*>(sW);
    for (int i = tid; i < NFRAG_HALVES * 2 / 16; i += 256) dstW[i] = src[i];
    const uint4* srcB = reinterpret_cast<const uint4*>(wsPack + NFRAG_HALVES);
    uint4*       dstB = reinterpret_cast<uint4*>(sB);
    for (int i = tid; i < NBIAS * 4 / 16; i += 256) dstB[i] = srcB[i];
  }

  // ---- Hash-grid encode: 128 points x 8 levels = 1024 tasks over 256 threads
  const int blockBase = blockIdx.x * 128;
  #pragma unroll
  for (int it = 0; it < 4; ++it) {
    int t   = tid + it * 256;
    int p   = t & 127;
    int lvl = t >> 7;
    int gp  = blockBase + p;
    if (gp < N) {
      // res = 16 * 64^(lvl/7)
      float res = exp2f((float)lvl * (6.0f / 7.0f)) * 16.0f;
      float px = points[gp * 3 + 0] * res;
      float py = points[gp * 3 + 1] * res;
      float pz = points[gp * 3 + 2] * res;
      float fx = floorf(px), fy = floorf(py), fz = floorf(pz);
      float rx = px - fx, ry = py - fy, rz = pz - fz;
      unsigned ix = (unsigned)(int)fx, iy = (unsigned)(int)fy, iz = (unsigned)(int)fz;
      const float4* tf = reinterpret_cast<const float4*>(tables) + (size_t)lvl * NGP_T;
      float a0 = 0.f, a1 = 0.f, a2 = 0.f, a3 = 0.f;
      #pragma unroll
      for (int c = 0; c < 8; ++c) {
        unsigned cx = (c >> 2) & 1u, cy = (c >> 1) & 1u, cz = c & 1u;
        unsigned h = (ix + cx) * 1u ^ (iy + cy) * 2654435761u ^ (iz + cz) * 805459861u;
        h &= (NGP_T - 1u);
        float4 f = tf[h];                                   // L2-resident gather
        float w = (cx ? rx : 1.f - rx) * (cy ? ry : 1.f - ry) * (cz ? rz : 1.f - rz);
        a0 += w * f.x; a1 += w * f.y; a2 += w * f.z; a3 += w * f.w;
      }
      int wv = p >> 4, m = p & 15;
      union { _Float16 h4[4]; uint2 u; } pk;
      pk.h4[0] = (_Float16)a0; pk.h4[1] = (_Float16)a1;
      pk.h4[2] = (_Float16)a2; pk.h4[3] = (_Float16)a3;
      *reinterpret_cast<uint2*>(&sAct[(wv * WAVE_ROWS + m) * ROW_H + 4 * lvl]) = pk.u;
    }
  }
  __syncthreads();

  // ---- Per-wave fused MLP over its 16-row tile ----
  const int lane = tid & 31;
  const int wave = tid >> 5;
  _Float16* act = sAct + wave * WAVE_ROWS * ROW_H;
  const int rowBase = blockBase + wave * 16;

  v8f c4[4];
  v8f c1[1];

  mlp_layer<1, 4>(act, sW, sB + 0,   0,  lane, c4);   // 32  -> 64, relu
  store_relu<4>(act, lane, c4);
  mlp_layer<2, 4>(act, sW, sB + 64,  4,  lane, c4);   // 64  -> 64, relu
  store_relu<4>(act, lane, c4);
  mlp_layer<2, 1>(act, sW, sB + 128, 12, lane, c1);   // 64  -> 16, linear

  // densities = relu(col 0); cols 1..15 shifted to act cols 0..14 (f16)
  {
    int col = lane & 15, rb = (lane >> 4) << 3;
    #pragma unroll
    for (int r = 0; r < 8; ++r) {
      int row = rb + r;
      float v = c1[0][r];
      if (col == 0) {
        int gp = rowBase + row;
        if (gp < N) out[(size_t)N * 3 + gp] = fmaxf(v, 0.0f);
      } else {
        act[row * ROW_H + (col - 1)] = (_Float16)v;
      }
    }
  }

  // SH-16 encode of view dirs -> act cols 15..30; col 31 zero-pad
  if (lane < 16) {
    int gp = rowBase + lane;
    float dx = 1.f, dy = 0.f, dz = 0.f;
    if (gp < N) { dx = views[gp * 3]; dy = views[gp * 3 + 1]; dz = views[gp * 3 + 2]; }
    float inv = rsqrtf(dx * dx + dy * dy + dz * dz);
    float x = dx * inv, y = dy * inv, z = dz * inv;
    float xx = x * x, yy = y * y, zz = z * z;
    float sh[16];
    sh[0]  = 0.28209479177387814f;
    sh[1]  = -0.48860251190291987f * y;
    sh[2]  = 0.48860251190291987f * z;
    sh[3]  = -0.48860251190291987f * x;
    sh[4]  = 1.0925484305920792f * x * y;
    sh[5]  = -1.0925484305920792f * y * z;
    sh[6]  = 0.94617469575755997f * zz - 0.31539156525252005f;
    sh[7]  = -1.0925484305920792f * x * z;
    sh[8]  = 0.54627421529603959f * (xx - yy);
    sh[9]  = -0.59004358992664352f * y * (3.0f * xx - yy);
    sh[10] = 2.8906114426405538f * x * y * z;
    sh[11] = -0.45704579946446572f * y * (4.0f * zz - xx - yy);
    sh[12] = 0.37317633259011546f * z * (2.0f * zz - 3.0f * xx - 3.0f * yy);
    sh[13] = -0.45704579946446572f * x * (4.0f * zz - xx - yy);
    sh[14] = 1.4453057213202769f * z * (xx - yy);
    sh[15] = -0.59004358992664352f * x * (xx - 3.0f * yy);
    _Float16* rp = act + lane * ROW_H;
    #pragma unroll
    for (int j = 0; j < 16; ++j) rp[15 + j] = (_Float16)sh[j];
    rp[31] = (_Float16)0.0f;
  }

  mlp_layer<1, 4>(act, sW, sB + 144, 14, lane, c4);   // 31(+pad) -> 64, relu
  store_relu<4>(act, lane, c4);
  mlp_layer<2, 4>(act, sW, sB + 208, 18, lane, c4);   // 64 -> 64, relu
  store_relu<4>(act, lane, c4);
  mlp_layer<2, 4>(act, sW, sB + 272, 26, lane, c4);   // 64 -> 64, relu
  store_relu<4>(act, lane, c4);
  mlp_layer<2, 1>(act, sW, sB + 336, 34, lane, c1);   // 64 -> 16 (3 used)

  // colors = sigmoid(cols 0..2)
  {
    int col = lane & 15, rb = (lane >> 4) << 3;
    if (col < 3) {
      #pragma unroll
      for (int r = 0; r < 8; ++r) {
        int gp = rowBase + rb + r;
        if (gp < N) {
          float v = c1[0][r];
          out[(size_t)gp * 3 + col] = 1.0f / (1.0f + __expf(-v));
        }
      }
    }
  }
}

// ---------------------------------------------------------------------------
extern "C" void kernel_launch(void* const* d_in, const int* in_sizes, int n_in,
                              void* d_out, int out_size, void* d_ws, size_t ws_size,
                              hipStream_t stream) {
  (void)n_in; (void)out_size; (void)ws_size;
  const float* points = (const float*)d_in[0];
  const float* views  = (const float*)d_in[1];
  const float* tables = (const float*)d_in[2];
  WPtrs P;
  for (int i = 0; i < 7; ++i) {
    P.W[i] = (const float*)d_in[3 + 2 * i];
    P.b[i] = (const float*)d_in[4 + 2 * i];
  }
  int N = in_sizes[0] / 3;

  _Float16* wsW = (_Float16*)d_ws;
  float*    wsB = (float*)((char*)d_ws + NFRAG_HALVES * 2);

  int packElems = NFRAG_HALVES + NBIAS;
  ngp_pack<<<(packElems + 255) / 256, 256, 0, stream>>>(P, wsW, wsB);

  int blocks = (N + 127) / 128;
  ngp_fwd<<<blocks, 256, 0, stream>>>(points, views, tables,
                                      (const _Float16*)d_ws, (float*)d_out, N);
}
